// DVE_loss_22239340659319
// MI455X (gfx1250) — compile-verified, matrix operand order checked by reading, not run on hardware
//
#include <hip/hip_runtime.h>
#include <hip/hip_bf16.h>
#include <math.h>

// ---------------------------------------------------------------------------
// DVE loss on MI455X (gfx1250, wave32, WMMA).
// B=8, N=1024, C=128.  All GEMMs via v_wmma_f32_16x16x32_f16.
// ---------------------------------------------------------------------------

#define BB 8
#define NN 1024
#define CC 128
#define SCALE_F 20.0f
#define SINK_TAU 0.3f
#define SINK_ITERS 30
#define EPSF 1e-12f

typedef __attribute__((ext_vector_type(16))) _Float16 v16h;
typedef __attribute__((ext_vector_type(8)))  float    v8f;

#define WMMA_F16(A, B, Cacc) \
    __builtin_amdgcn_wmma_f32_16x16x32_f16(false, (A), false, (B), (short)0, (Cacc), false, false)

// ---------------------------------------------------------------------------
// Fragment loader per CDNA5 ISA 7.12.2 (16-bit A-matrix 16x32 layout).
// lane 0..15 : row = base + lane, K-group 0 ; lane 16..31 : same rows, K-group 1
// VGPR v<4   : K = 8*kg + 2v, 2v+1 ; VGPR v>=4 : K = 16 + 8*kg + 2(v-4), +1
// ---------------------------------------------------------------------------
static __device__ __forceinline__ int frag_koff(int v, int kg) {
    return (v < 4) ? (8 * kg + 2 * v) : (16 + 8 * kg + 2 * (v - 4));
}

// rows of `p` contiguous (row-major, leading dim ld): element (r, k) = p[r*ld + k]
static __device__ __forceinline__ v16h load_frag_rowmajor(const _Float16* __restrict__ p,
                                                          int ld, int rowBase, int kBase,
                                                          int lane) {
    union { v16h v; unsigned u[8]; } r;
    int row = rowBase + (lane & 15);
    int kg  = (lane >> 4) & 1;
    const _Float16* rp = p + (size_t)row * ld + kBase;
#pragma unroll
    for (int v = 0; v < 8; ++v) {
        int k0 = frag_koff(v, kg);
        r.u[v] = *(const unsigned*)(rp + k0);   // two contiguous f16 (4B aligned)
    }
    return r.v;
}

// ---------------------------------------------------------------------------
// K0: L2-normalize feats rows, scale by 20, split even/odd batches -> f16
// ---------------------------------------------------------------------------
__global__ void k_normalize(const float* __restrict__ feats,
                            _Float16* __restrict__ f1h, _Float16* __restrict__ f2h) {
    int g   = blockIdx.x;          // bin*1024 + n
    int bin = g >> 10;
    int n   = g & (NN - 1);
    int t   = threadIdx.x;         // 0..127
    const float* p = feats + (size_t)g * CC;
    float v  = p[t];
    float ss = v * v;
    __shared__ float sm[4];
#pragma unroll
    for (int o = 16; o > 0; o >>= 1) ss += __shfl_down(ss, o);
    int lane = t & 31, wid = t >> 5;
    if (lane == 0) sm[wid] = ss;
    __syncthreads();
    __shared__ float stot;
    if (t == 0) stot = sm[0] + sm[1] + sm[2] + sm[3];
    __syncthreads();
    float scale = SCALE_F / fmaxf(sqrtf(stot), EPSF);
    _Float16* dst = (bin & 1) ? f2h : f1h;
    dst[((size_t)(bin >> 1) * NN + n) * CC + t] = (_Float16)(v * scale);
}

// ---------------------------------------------------------------------------
// K0b: transpose f1h [8][1024][128] -> f1hT [8][128][1024] (for k_via B-frags)
// ---------------------------------------------------------------------------
__global__ void k_transpose(const _Float16* __restrict__ src, _Float16* __restrict__ dst) {
    int g = blockIdx.x;            // b*1024 + n
    int b = g >> 10, n = g & (NN - 1);
    int t = threadIdx.x;           // 0..127 (channel)
    dst[((size_t)b * CC + t) * NN + n] = src[(size_t)g * CC + t];
}

// ---------------------------------------------------------------------------
// K1: C[b] = A[b] (NxC) x Bm[(b+rollB)%8]^T (NxC) -> NxN fp32
// One wave per 16x64 tile: A-fragment reused across 4 B tiles, 16 WMMAs.
// grid = (N/16, N/64, B), block = 32
// ---------------------------------------------------------------------------
__global__ void k_gemm_corr(const _Float16* __restrict__ A, const _Float16* __restrict__ Bm,
                            float* __restrict__ C, int rollB) {
    int tn = blockIdx.x, tm4 = blockIdx.y, b = blockIdx.z;
    int bB = (b + rollB) & (BB - 1);
    int lane = threadIdx.x;
    const _Float16* Ab = A  + (size_t)b  * NN * CC;
    const _Float16* Bb = Bm + (size_t)bB * NN * CC;
    v8f acc0 = {}, acc1 = {}, acc2 = {}, acc3 = {};
#pragma unroll
    for (int kk = 0; kk < CC; kk += 32) {
        v16h af = load_frag_rowmajor(Ab, CC, tn * 16, kk, lane);
        v16h b0 = load_frag_rowmajor(Bb, CC, tm4 * 64 +  0, kk, lane);
        acc0 = WMMA_F16(af, b0, acc0);
        v16h b1 = load_frag_rowmajor(Bb, CC, tm4 * 64 + 16, kk, lane);
        acc1 = WMMA_F16(af, b1, acc1);
        v16h b2 = load_frag_rowmajor(Bb, CC, tm4 * 64 + 32, kk, lane);
        acc2 = WMMA_F16(af, b2, acc2);
        v16h b3 = load_frag_rowmajor(Bb, CC, tm4 * 64 + 48, kk, lane);
        acc3 = WMMA_F16(af, b3, acc3);
    }
    int colb  = tm4 * 64 + (lane & 15);
    int mrow0 = tn * 16 + (((lane >> 4) & 1) << 3);
    float* Cb = C + (size_t)b * NN * NN + (size_t)mrow0 * NN + colb;
#pragma unroll
    for (int r = 0; r < 8; ++r) {
        Cb[(size_t)r * NN +  0] = acc0[r];
        Cb[(size_t)r * NN + 16] = acc1[r];
        Cb[(size_t)r * NN + 32] = acc2[r];
        Cb[(size_t)r * NN + 48] = acc3[r];
    }
}

// ---------------------------------------------------------------------------
// K2: per-row stats of an [B*N, N] matrix: lse = max + log(sum exp(x-max)).
// Optionally argmax -> match flag (argmax == row index within batch).
// ---------------------------------------------------------------------------
__global__ void k_rowstats(const float* __restrict__ C, float* __restrict__ lse,
                           float* __restrict__ match, int doArgmax) {
    int row = blockIdx.x;
    const float* p = C + (size_t)row * NN;
    int t = threadIdx.x, lane = t & 31, wid = t >> 5;
    __shared__ float smax[8]; __shared__ int simax[8]; __shared__ float ssum[8];
    __shared__ float rmax_s; __shared__ int rarg_s;

    float vmax = -3.4e38f; int imax = 0;
    for (int m = t; m < NN; m += 256) {
        float v = p[m];
        if (v > vmax || (v == vmax && m < imax)) { vmax = v; imax = m; }
    }
#pragma unroll
    for (int o = 16; o > 0; o >>= 1) {
        float ov = __shfl_down(vmax, o);
        int   oi = __shfl_down(imax, o);
        if (ov > vmax || (ov == vmax && oi < imax)) { vmax = ov; imax = oi; }
    }
    if (lane == 0) { smax[wid] = vmax; simax[wid] = imax; }
    __syncthreads();
    if (t == 0) {
        float mv = smax[0]; int mi = simax[0];
        for (int w = 1; w < 8; ++w)
            if (smax[w] > mv || (smax[w] == mv && simax[w] < mi)) { mv = smax[w]; mi = simax[w]; }
        rmax_s = mv; rarg_s = mi;
    }
    __syncthreads();
    float rmax = rmax_s;
    float s = 0.f;
    for (int m = t; m < NN; m += 256) s += __expf(p[m] - rmax);
#pragma unroll
    for (int o = 16; o > 0; o >>= 1) s += __shfl_down(s, o);
    if (lane == 0) ssum[wid] = s;
    __syncthreads();
    if (t == 0) {
        float tot = 0.f;
        for (int w = 0; w < 8; ++w) tot += ssum[w];
        lse[row] = rmax + logf(tot);
        if (doArgmax) match[row] = (rarg_s == (row & (NN - 1))) ? 1.f : 0.f;
    }
}

// ---------------------------------------------------------------------------
// K3: via[b] = softmax(C1[b]) x fa[b]  (fa = f1[(b+1)%8]), f16 output.
// One wave per 16-row block produces the full 16x128 output: 8 accumulators,
// exp A-fragment built ONCE per m-step and reused for 8 WMMAs (256 WMMAs/wave).
// B-fragments from pre-transposed f1hT (contiguous-pair gathers -> b128 loads).
// grid = (N/16, B), block = 32
// ---------------------------------------------------------------------------
__global__ void k_via(const float* __restrict__ C1, const float* __restrict__ lse1,
                      const _Float16* __restrict__ f1hT, _Float16* __restrict__ viah) {
    int tn = blockIdx.x, b = blockIdx.y;
    int bB = (b + 1) & (BB - 1);
    int lane = threadIdx.x;
    int kg  = (lane >> 4) & 1;
    int row = tn * 16 + (lane & 15);
    const float* rowp = C1 + (size_t)b * NN * NN + (size_t)row * NN;
    const _Float16* faT = f1hT + (size_t)bB * CC * NN;   // [128][1024]
    float l = lse1[b * NN + row];

    v8f acc[8];
#pragma unroll
    for (int tc = 0; tc < 8; ++tc) acc[tc] = (v8f){};

    for (int m0 = 0; m0 < NN; m0 += 32) {
        union { v16h v; _Float16 h[16]; } a;
#pragma unroll
        for (int v = 0; v < 8; ++v) {
            int k0 = frag_koff(v, kg);
            a.h[2 * v]     = (_Float16)__expf(rowp[m0 + k0]     - l);
            a.h[2 * v + 1] = (_Float16)__expf(rowp[m0 + k0 + 1] - l);
        }
#pragma unroll
        for (int tc = 0; tc < 8; ++tc) {
            v16h bf = load_frag_rowmajor(faT, NN, tc * 16, m0, lane);
            acc[tc] = WMMA_F16(a.v, bf, acc[tc]);
        }
    }
    int mrow0 = tn * 16 + (kg << 3);
    _Float16* Vb = viah + (size_t)b * NN * CC + (size_t)mrow0 * CC + (lane & 15);
#pragma unroll
    for (int tc = 0; tc < 8; ++tc)
#pragma unroll
        for (int r = 0; r < 8; ++r)
            Vb[(size_t)r * CC + tc * 16] = (_Float16)acc[tc][r];
}

// ---------------------------------------------------------------------------
// Sinkhorn kernels (log-space, in place, L2-resident).
// ---------------------------------------------------------------------------
__global__ void k_scale_la(const float* __restrict__ C2, float* __restrict__ LA) {
    size_t total = (size_t)BB * NN * NN;
    size_t i = (size_t)blockIdx.x * blockDim.x + threadIdx.x;
    size_t stride = (size_t)gridDim.x * blockDim.x;
    for (; i < total; i += stride) LA[i] = C2[i] * (1.0f / SINK_TAU);
}

__global__ void k_sink_row(float* __restrict__ LA) {
    int row = blockIdx.x;
    float* p = LA + (size_t)row * NN;
    int t = threadIdx.x, lane = t & 31, wid = t >> 5;
    __shared__ float sm[8]; __shared__ float bcast;
    float vmax = -3.4e38f;
    for (int m = t; m < NN; m += 256) vmax = fmaxf(vmax, p[m]);
#pragma unroll
    for (int o = 16; o > 0; o >>= 1) vmax = fmaxf(vmax, __shfl_down(vmax, o));
    if (lane == 0) sm[wid] = vmax;
    __syncthreads();
    if (t == 0) {
        float mv = sm[0];
        for (int w = 1; w < 8; ++w) mv = fmaxf(mv, sm[w]);
        bcast = mv;
    }
    __syncthreads();
    float rmax = bcast;
    float s = 0.f;
    for (int m = t; m < NN; m += 256) s += __expf(p[m] - rmax);
#pragma unroll
    for (int o = 16; o > 0; o >>= 1) s += __shfl_down(s, o);
    if (lane == 0) sm[wid] = s;
    __syncthreads();
    if (t == 0) {
        float tot = 0.f;
        for (int w = 0; w < 8; ++w) tot += sm[w];
        bcast = rmax + logf(tot);
    }
    __syncthreads();
    float l = bcast;
    for (int m = t; m < NN; m += 256) p[m] -= l;
}

__global__ void k_sink_col(float* __restrict__ LA) {
    int b = blockIdx.x >> 10;
    int col = blockIdx.x & (NN - 1);
    float* p = LA + (size_t)b * NN * NN + col;
    int t = threadIdx.x, lane = t & 31, wid = t >> 5;
    __shared__ float sm[8]; __shared__ float bcast;
    float vmax = -3.4e38f;
    for (int i = t; i < NN; i += 256) vmax = fmaxf(vmax, p[(size_t)i * NN]);
#pragma unroll
    for (int o = 16; o > 0; o >>= 1) vmax = fmaxf(vmax, __shfl_down(vmax, o));
    if (lane == 0) sm[wid] = vmax;
    __syncthreads();
    if (t == 0) {
        float mv = sm[0];
        for (int w = 1; w < 8; ++w) mv = fmaxf(mv, sm[w]);
        bcast = mv;
    }
    __syncthreads();
    float cmax = bcast;
    float s = 0.f;
    for (int i = t; i < NN; i += 256) s += __expf(p[(size_t)i * NN] - cmax);
#pragma unroll
    for (int o = 16; o > 0; o >>= 1) s += __shfl_down(s, o);
    if (lane == 0) sm[wid] = s;
    __syncthreads();
    if (t == 0) {
        float tot = 0.f;
        for (int w = 0; w < 8; ++w) tot += sm[w];
        bcast = cmax + logf(tot);
    }
    __syncthreads();
    float l = bcast;
    for (int i = t; i < NN; i += 256) p[(size_t)i * NN] -= l;
}

// ---------------------------------------------------------------------------
// K4: per-row loss partials. One block per (b, n) row. Deterministic
// (per-row partials, no float atomics), reduced by k_finalize.
// ---------------------------------------------------------------------------
__global__ void k_loss(const float* __restrict__ C2, const float* __restrict__ lse2,
                       const float* __restrict__ C3, const float* __restrict__ lse3,
                       const float* __restrict__ LA, const float* __restrict__ pc0,
                       float* __restrict__ partials) {
    int row = blockIdx.x;                    // b*N + n
    int n = row & (NN - 1);
    const float* p2 = C2 + (size_t)row * NN;
    const float* p3 = C3 + (size_t)row * NN;
    const float* pl = LA + (size_t)row * NN;
    const float* pc = pc0 + ((size_t)(row >> 10) * NN) * 3;
    float l2r = lse2[row], l3r = lse3[row];
    float p0x = pc[(size_t)n * 3 + 0];
    float p0y = pc[(size_t)n * 3 + 1];
    float p0z = pc[(size_t)n * 3 + 2];
    int t = threadIdx.x, lane = t & 31, wid = t >> 5;
    float sl = 0.f, sc = 0.f, sp = 0.f;
    for (int m = t; m < NN; m += 256) {
        float dx = p0x - pc[(size_t)m * 3 + 0];
        float dy = p0y - pc[(size_t)m * 3 + 1];
        float dz = p0z - pc[(size_t)m * 3 + 2];
        float d2 = dx * dx + dy * dy + dz * dz;
        float dist = sqrtf(sqrtf(d2));       // (||diff||)^0.5 = d2^0.25
        float sm1a2 = __expf(p2[m] - l2r);
        float sm12  = __expf(p3[m] - l3r);
        float sink  = __expf(pl[m]);
        sl += dist * (sm1a2 + sm12);
        sc += fabsf(sink - sm1a2);
        sp += fabsf(((m == n) ? 1.f : 0.f) - sink);
    }
    __shared__ float sh[8];
#pragma unroll
    for (int o = 16; o > 0; o >>= 1) sl += __shfl_down(sl, o);
    if (lane == 0) sh[wid] = sl;
    __syncthreads();
    if (t == 0) { float v = 0; for (int w = 0; w < 8; ++w) v += sh[w]; partials[row] = v; }
    __syncthreads();
#pragma unroll
    for (int o = 16; o > 0; o >>= 1) sc += __shfl_down(sc, o);
    if (lane == 0) sh[wid] = sc;
    __syncthreads();
    if (t == 0) { float v = 0; for (int w = 0; w < 8; ++w) v += sh[w]; partials[BB * NN + row] = v; }
    __syncthreads();
#pragma unroll
    for (int o = 16; o > 0; o >>= 1) sp += __shfl_down(sp, o);
    if (lane == 0) sh[wid] = sp;
    __syncthreads();
    if (t == 0) { float v = 0; for (int w = 0; w < 8; ++w) v += sh[w]; partials[2 * BB * NN + row] = v; }
}

__global__ void k_finalize(const float* __restrict__ partials,
                           const float* __restrict__ match, float* __restrict__ out) {
    int t = threadIdx.x, lane = t & 31, wid = t >> 5;
    float sl = 0.f, sc = 0.f, sp = 0.f, smch = 0.f;
    for (int i = t; i < BB * NN; i += 256) {
        sl   += partials[i];
        sc   += partials[BB * NN + i];
        sp   += partials[2 * BB * NN + i];
        smch += match[i];
    }
    __shared__ float sh[8 * 4];
#pragma unroll
    for (int o = 16; o > 0; o >>= 1) {
        sl   += __shfl_down(sl, o);
        sc   += __shfl_down(sc, o);
        sp   += __shfl_down(sp, o);
        smch += __shfl_down(smch, o);
    }
    if (lane == 0) { sh[wid] = sl; sh[8 + wid] = sc; sh[16 + wid] = sp; sh[24 + wid] = smch; }
    __syncthreads();
    if (t == 0) {
        float tl = 0, tc = 0, tp = 0, tm = 0;
        for (int w = 0; w < 8; ++w) { tl += sh[w]; tc += sh[8 + w]; tp += sh[16 + w]; tm += sh[24 + w]; }
        const float invN = 1.0f / (float)NN, invB = 1.0f / (float)BB;
        float loss = tl * invN * invB;
        float Lc   = 3.0f * tc * invN * invB;
        float perm = 3.0f * tp * invN * invB;
        float mtch = tm * invB;
        out[0] = loss + Lc;   // total_loss / B (lambda_lc = 1)
        out[1] = loss;
        out[2] = Lc;
        out[3] = mtch;
        out[4] = perm;
    }
}

// ---------------------------------------------------------------------------
// Launcher
// ---------------------------------------------------------------------------
extern "C" void kernel_launch(void* const* d_in, const int* in_sizes, int n_in,
                              void* d_out, int out_size, void* d_ws, size_t ws_size,
                              hipStream_t stream) {
    const float* feats = (const float*)d_in[0];   // [16,1024,128] f32
    const float* pc0   = (const float*)d_in[1];   // [8,1024,3]  f32
    float* out = (float*)d_out;                   // 5 scalars

    // workspace layout (256B aligned chunks)
    char* ws = (char*)d_ws;
    size_t off = 0;
    auto take = [&](size_t bytes) -> char* {
        char* p = ws + off;
        off = (off + bytes + 255) & ~(size_t)255;
        return p;
    };
    const size_t featsH = (size_t)BB * NN * CC * sizeof(_Float16);   // 2 MB
    const size_t matF   = (size_t)BB * NN * NN * sizeof(float);      // 32 MB
    _Float16* f1h  = (_Float16*)take(featsH);
    _Float16* f2h  = (_Float16*)take(featsH);
    _Float16* f1hT = (_Float16*)take(featsH);
    _Float16* viah = (_Float16*)take(featsH);
    float* C1   = (float*)take(matF);         // corr_1a; later aliased by LA
    float* C2   = (float*)take(matF);         // corr_1a2
    float* C3   = (float*)take(matF);         // corr_12
    float* lse1 = (float*)take(BB * NN * sizeof(float));
    float* lse2 = (float*)take(BB * NN * sizeof(float));
    float* lse3 = (float*)take(BB * NN * sizeof(float));
    float* matc = (float*)take(BB * NN * sizeof(float));
    float* part = (float*)take(3 * BB * NN * sizeof(float));
    float* LA = C1;                           // C1 dead after k_via -> reuse

    // 1) normalize + f16 convert, then transpose f1 for the via-GEMM B operand
    k_normalize<<<dim3(2 * BB * NN), dim3(CC), 0, stream>>>(feats, f1h, f2h);
    k_transpose<<<dim3(BB * NN), dim3(CC), 0, stream>>>(f1h, f1hT);

    // 2) corr_1a = f1 x roll(f1,-1)^T   (WMMA, 16x64 tiles)
    k_gemm_corr<<<dim3(NN / 16, NN / 64, BB), dim3(32), 0, stream>>>(f1h, f1h, C1, 1);
    k_rowstats<<<dim3(BB * NN), dim3(256), 0, stream>>>(C1, lse1, matc, 0);

    // 3) f1_via_fa = softmax(corr_1a) x fa   (WMMA, full 16x128 per wave)
    k_via<<<dim3(NN / 16, BB), dim3(32), 0, stream>>>(C1, lse1, f1hT, viah);

    // 4) corr_1a2 = via x f2^T ; corr_12 = f1 x f2^T   (WMMA)
    k_gemm_corr<<<dim3(NN / 16, NN / 64, BB), dim3(32), 0, stream>>>(viah, f2h, C2, 0);
    k_gemm_corr<<<dim3(NN / 16, NN / 64, BB), dim3(32), 0, stream>>>(f1h, f2h, C3, 0);

    // 5) row stats (lse + argmax/match for corr_1a2)
    k_rowstats<<<dim3(BB * NN), dim3(256), 0, stream>>>(C2, lse2, matc, 1);
    k_rowstats<<<dim3(BB * NN), dim3(256), 0, stream>>>(C3, lse3, matc, 0);

    // 6) Sinkhorn: LA = C2 / tau; 30x (row-normalize, col-normalize), log space
    k_scale_la<<<dim3(8192), dim3(256), 0, stream>>>(C2, LA);
    for (int it = 0; it < SINK_ITERS; ++it) {
        k_sink_row<<<dim3(BB * NN), dim3(256), 0, stream>>>(LA);
        k_sink_col<<<dim3(BB * NN), dim3(256), 0, stream>>>(LA);
    }

    // 7) fused loss partials + deterministic final reduction
    k_loss<<<dim3(BB * NN), dim3(256), 0, stream>>>(C2, lse2, C3, lse3, LA, pc0, part);
    k_finalize<<<dim3(1), dim3(256), 0, stream>>>(part, matc, out);

    (void)in_sizes; (void)n_in; (void)out_size; (void)ws_size;
}
